// GATCritic_26723286516180
// MI455X (gfx1250) — matrix operand order repaired and checked by reading, not compiled, not run on hardware
//
#include <hip/hip_runtime.h>
#include <hip/hip_bf16.h>

// ---------------------------------------------------------------------------
// GAT critic for MI455X (gfx1250, wave32).
// Heavy GEMMs (H @ W, [32768x512]x[512x512]) run on v_wmma_f32_16x16x32_bf16
// with f32 accumulation. Operands are pre-converted to bf16 once per layer
// (weights transposed), and GEMM tiles are staged into LDS with
// GLOBAL_LOAD_ASYNC_TO_LDS_B128 (ASYNCcnt) when the toolchain exposes it.
// Attention segment-max/segment-sum use L2 atomics.
// ---------------------------------------------------------------------------

#define Gg   512
#define Pp   64
#define Nn   (Gg * Pp)        // 32768 nodes
#define DEG  8
#define Ee   (Nn * DEG)       // 262144 edges
#define HID  64
#define HEADS 8
#define HC   (HID * HEADS)    // 512
#define NA   5
#define AD   3
#define INDIM (HC * (NA + 1) + NA * AD)  // 3087
#define FHID (HID * 4)        // 256
#define NEG_SLOPE 0.2f

typedef __attribute__((ext_vector_type(16))) __bf16 v16bf;
typedef __attribute__((ext_vector_type(8)))  float  v8f;
typedef __attribute__((ext_vector_type(4)))  unsigned int v4u;
typedef __attribute__((ext_vector_type(4)))  int v4i;

// pointer types matching the async-to-LDS builtin prototype:
// (int4 addrspace(1)*, int4 addrspace(3)*, imm offset, imm cpol)
typedef __attribute__((address_space(1))) v4i* g_v4i_ptr;
typedef __attribute__((address_space(3))) v4i* l_v4i_ptr;

union ABfrag { v16bf f; v4u q[2]; };

#if defined(__has_builtin)
#  if __has_builtin(__builtin_amdgcn_global_load_async_to_lds_b128)
#    define HAVE_ASYNC_LDS 1
#  endif
#  if __has_builtin(__builtin_amdgcn_s_wait_asynccnt)
#    define HAVE_WAIT_ASYNC_BUILTIN 1
#  endif
#endif
#ifndef HAVE_ASYNC_LDS
#  define HAVE_ASYNC_LDS 0
#endif
#ifndef HAVE_WAIT_ASYNC_BUILTIN
#  define HAVE_WAIT_ASYNC_BUILTIN 0
#endif

// stage one 16-byte chunk global(bf16) -> LDS
static __device__ __forceinline__ void stage16(const unsigned short* g, unsigned short* l) {
#if HAVE_ASYNC_LDS
    __builtin_amdgcn_global_load_async_to_lds_b128((g_v4i_ptr)g, (l_v4i_ptr)l, 0, 0);
#else
    *(v4u*)l = *(const v4u*)g;
#endif
}
static __device__ __forceinline__ void stage_wait() {
#if HAVE_ASYNC_LDS
#  if HAVE_WAIT_ASYNC_BUILTIN
    __builtin_amdgcn_s_wait_asynccnt(0);
#  else
    asm volatile("s_wait_asynccnt 0x0" ::: "memory");
#  endif
#endif
}

static __device__ __forceinline__ unsigned short f32_to_bf16_rne(float x) {
    unsigned u = __float_as_uint(x);
    unsigned rounding = 0x7FFFu + ((u >> 16) & 1u);
    return (unsigned short)((u + rounding) >> 16);
}

// order-preserving float <-> uint encoding for atomic segment-max
static __device__ __forceinline__ unsigned enc_f32(float f) {
    unsigned u = __float_as_uint(f);
    return (u >> 31) ? ~u : (u | 0x80000000u);
}
static __device__ __forceinline__ float dec_f32(unsigned e) {
    return (e & 0x80000000u) ? __uint_as_float(e ^ 0x80000000u)
                             : __uint_as_float(~e);
}
static __device__ __forceinline__ float leaky(float a) {
    return a > 0.0f ? a : NEG_SLOPE * a;
}

// ---------------------------------------------------------------------------
// fills
// ---------------------------------------------------------------------------
__global__ void fill_f32_kernel(float* p, float v, int n) {
    int i = blockIdx.x * blockDim.x + threadIdx.x;
    if (i < n) p[i] = v;
}
__global__ void fill_u32_kernel(unsigned* p, unsigned v, int n) {
    int i = blockIdx.x * blockDim.x + threadIdx.x;
    if (i < n) p[i] = v;
}

// ---------------------------------------------------------------------------
// ce[h] = sum_c We[h*64+c] * ae[h*64+c]   (rank-1 edge projection)
// ---------------------------------------------------------------------------
__global__ void ce_kernel(const float* __restrict__ We, const float* __restrict__ ae,
                          float* __restrict__ ce) {
    int h = threadIdx.x;
    if (h >= HEADS) return;
    float s = 0.0f;
    for (int c = 0; c < HID; ++c) s += We[h * HID + c] * ae[h * HID + c];
    ce[h] = s;
}

// ---------------------------------------------------------------------------
// Layer-1 transform: T = x @ W1, K = 4 (too thin for WMMA; VALU)
// ---------------------------------------------------------------------------
__global__ void lin1_kernel(const float* __restrict__ x, const float* __restrict__ W1,
                            float* __restrict__ T) {
    int idx = blockIdx.x * blockDim.x + threadIdx.x;
    if (idx >= Nn * HC) return;
    int n = idx >> 9, j = idx & (HC - 1);
    float s = 0.0f;
#pragma unroll
    for (int k = 0; k < 4; ++k) s += x[n * 4 + k] * W1[k * HC + j];
    T[idx] = s;
}

// ---------------------------------------------------------------------------
// W [512x512 f32, row-major (k,n)] -> Wt [bf16, transposed: Wt[n*512+k]]
// ---------------------------------------------------------------------------
__global__ void wtcvt_kernel(const float* __restrict__ W, unsigned short* __restrict__ Wt) {
    int idx = blockIdx.x * blockDim.x + threadIdx.x;
    if (idx >= HC * HC) return;
    int n = idx >> 9, k = idx & (HC - 1);
    Wt[idx] = f32_to_bf16_rne(W[k * HC + n]);
}

// ---------------------------------------------------------------------------
// WMMA bf16 GEMM:  C[M x 512] = A[M x 512] @ W[512 x 512]
// A pre-converted bf16 row-major; W pre-converted bf16 TRANSPOSED ([n][k]).
// block = 256 threads (8 waves); block tile 64(M) x 128(N); K-step 32.
// Wave (w) computes a 16x64 strip: rows (w&3)*16, cols (w>>2)*64, 4 C tiles.
// Tiles staged to LDS with async b128 copies (ASYNCcnt) when available;
// fragments built with two contiguous 16B ds reads per lane matching the
// ISA 16-bit A-16x32 / B-32x16 VGPR layouts.
// ---------------------------------------------------------------------------
__global__ __launch_bounds__(256) void gemm_bf16_wmma_kernel(
    const unsigned short* __restrict__ Abf, const unsigned short* __restrict__ Wt,
    float* __restrict__ C) {
    __shared__ __align__(16) unsigned short As[64][32];   // 4 KB
    __shared__ __align__(16) unsigned short Bs[128][32];  // 8 KB ([n][k])

    const int tid  = threadIdx.x;
    const int wave = tid >> 5;
    const int lane = tid & 31;
    const int m_blk = blockIdx.x * 64;
    const int n_blk = blockIdx.y * 128;
    const int wm = (wave & 3) * 16;   // row strip within block
    const int wn = (wave >> 2) * 64;  // col half within block

    // staging coordinates: each thread owns one 16B chunk of A, two of B
    const int ar = tid >> 2, ac = (tid & 3) * 8;           // A: 64 rows x 4 chunks
    const int b0 = tid, b1 = tid + 256;                    // B: 128 rows x 4 chunks

    v8f acc[4] = {};

    for (int kb = 0; kb < HC; kb += 32) {
        stage16(&Abf[(size_t)(m_blk + ar) * HC + kb + ac], &As[ar][ac]);
        {
            int n0 = b0 >> 2, c0 = (b0 & 3) * 8;
            stage16(&Wt[(size_t)(n_blk + n0) * HC + kb + c0], &Bs[n0][c0]);
            int n1 = b1 >> 2, c1 = (b1 & 3) * 8;
            stage16(&Wt[(size_t)(n_blk + n1) * HC + kb + c1], &Bs[n1][c1]);
        }
        stage_wait();
        __syncthreads();

        const int sel = lane >> 4;           // half-wave select
        const int r   = wm + (lane & 15);    // A row for this lane
        ABfrag af;
        // lanes 0-15: K 0-7 & 16-23 ; lanes 16-31: K 8-15 & 24-31
        af.q[0] = *(const v4u*)&As[r][sel * 8];
        af.q[1] = *(const v4u*)&As[r][16 + sel * 8];

#pragma unroll
        for (int t = 0; t < 4; ++t) {
            const int nn = wn + t * 16 + (lane & 15);
            ABfrag bf;
            // lanes 0-15: K 0-7 & 16-23 of col nn ; lanes 16-31: K 8-15 & 24-31
            bf.q[0] = *(const v4u*)&Bs[nn][sel * 8];
            bf.q[1] = *(const v4u*)&Bs[nn][16 + sel * 8];
            acc[t] = __builtin_amdgcn_wmma_f32_16x16x32_bf16(
                false, af.f, false, bf.f, (short)0, acc[t], false, false);
        }
        __syncthreads();
    }

    // C/D layout: VGPR i -> (M=i, N=lane) for lanes 0-15, (M=i+8, N=lane-16)
    const int col  = n_blk + wn + (lane & 15);
    const int row0 = m_blk + wm + (lane >> 4) * 8;
#pragma unroll
    for (int t = 0; t < 4; ++t) {
#pragma unroll
        for (int i = 0; i < 8; ++i)
            C[(size_t)(row0 + i) * HC + (col + t * 16)] = acc[t][i];
    }
}

// ---------------------------------------------------------------------------
// Node attention projections: als[n,h], ald[n,h]
// ---------------------------------------------------------------------------
__global__ void nodeproj_kernel(const float* __restrict__ T,
                                const float* __restrict__ a_s, const float* __restrict__ a_d,
                                float* __restrict__ als, float* __restrict__ ald) {
    int idx = blockIdx.x * blockDim.x + threadIdx.x;
    if (idx >= Nn * HEADS) return;
    int n = idx >> 3, h = idx & 7;
    const float* row = T + (size_t)n * HC + h * HID;
    float s = 0.0f, d = 0.0f;
    for (int c = 0; c < HID; ++c) {
        float v = row[c];
        s += v * a_s[h * HID + c];
        d += v * a_d[h * HID + c];
    }
    als[idx] = s;
    ald[idx] = d;
}

// ---------------------------------------------------------------------------
// Degree + summed edge_attr per dst (for self-loop fill_value='mean')
// ---------------------------------------------------------------------------
__global__ void deg_kernel(const int* __restrict__ ei, const float* __restrict__ attr,
                           float* __restrict__ cnt, float* __restrict__ sattr) {
    int e = blockIdx.x * blockDim.x + threadIdx.x;
    if (e >= Ee) return;
    int d = ei[Ee + e];
    atomicAdd(&cnt[d], 1.0f);
    atomicAdd(&sattr[d], attr[e]);
}

// ---------------------------------------------------------------------------
// Per-edge attention logits + segment-max via encoded atomicMax
// ---------------------------------------------------------------------------
__global__ void edge_alpha_kernel(const int* __restrict__ ei, const float* __restrict__ attr,
                                  const float* __restrict__ als, const float* __restrict__ ald,
                                  const float* __restrict__ ce,
                                  float* __restrict__ alpha, unsigned* __restrict__ mmax) {
    int idx = blockIdx.x * blockDim.x + threadIdx.x;
    if (idx >= Ee * HEADS) return;
    int e = idx >> 3, h = idx & 7;
    int s = ei[e], d = ei[Ee + e];
    float a = leaky(als[s * HEADS + h] + ald[d * HEADS + h] + ce[h] * attr[e]);
    alpha[idx] = a;
    atomicMax(&mmax[d * HEADS + h], enc_f32(a));
}

__global__ void self_alpha_kernel(const float* __restrict__ als, const float* __restrict__ ald,
                                  const float* __restrict__ ce,
                                  const float* __restrict__ cnt, const float* __restrict__ sattr,
                                  float* __restrict__ alpha_self, unsigned* __restrict__ mmax) {
    int idx = blockIdx.x * blockDim.x + threadIdx.x;
    if (idx >= Nn * HEADS) return;
    int n = idx >> 3, h = idx & 7;
    float la = sattr[n] / fmaxf(cnt[n], 1.0f);
    float a = leaky(als[idx] + ald[idx] + ce[h] * la);
    alpha_self[idx] = a;
    atomicMax(&mmax[n * HEADS + h], enc_f32(a));
}

// ---------------------------------------------------------------------------
// Pass 2: ex = exp(alpha - max); denom += ex; acc[dst,:] += ex * T[src,:]
// one block (64 threads) per edge
// ---------------------------------------------------------------------------
__global__ __launch_bounds__(64) void edge_acc_kernel(
    const int* __restrict__ ei, const float* __restrict__ alpha,
    const unsigned* __restrict__ mmax, const float* __restrict__ T,
    float* __restrict__ denom, float* __restrict__ acc) {
    __shared__ float ex[HEADS];
    int e = blockIdx.x;
    int t = threadIdx.x;
    int s = ei[e], d = ei[Ee + e];
    if (t < HEADS) {
        float a  = alpha[e * HEADS + t];
        float mm = dec_f32(mmax[d * HEADS + t]);
        float x  = __expf(a - mm);
        ex[t] = x;
        atomicAdd(&denom[d * HEADS + t], x);
    }
    __syncthreads();
    const float* src = T + (size_t)s * HC;
    float* dst = acc + (size_t)d * HC;
    for (int c = t; c < HC; c += 64)
        atomicAdd(&dst[c], ex[c >> 6] * src[c]);
}

// self-loop pass 2a: convert alpha_self -> ex in place, bump denom
__global__ void self_exp_kernel(float* __restrict__ alpha_self,
                                const unsigned* __restrict__ mmax,
                                float* __restrict__ denom) {
    int idx = blockIdx.x * blockDim.x + threadIdx.x;
    if (idx >= Nn * HEADS) return;
    float x = __expf(alpha_self[idx] - dec_f32(mmax[idx]));
    alpha_self[idx] = x;
    denom[idx] += x;   // exclusive per idx; stream-ordered after edge_acc
}

// self-loop pass 2b: acc[n,:] += ex * T[n,:]
__global__ void self_acc_kernel(const float* __restrict__ exs, const float* __restrict__ T,
                                float* __restrict__ acc) {
    int idx = blockIdx.x * blockDim.x + threadIdx.x;
    if (idx >= Nn * HC) return;
    int n = idx >> 9, c = idx & (HC - 1);
    acc[idx] += exs[n * HEADS + (c >> 6)] * T[idx];
}

// ---------------------------------------------------------------------------
// finalize: out = relu(acc / (denom + 1e-16) + b)   (in place on acc),
// plus fused bf16 copy feeding the next layer's WMMA GEMM.
// ---------------------------------------------------------------------------
__global__ void finalize_kernel(float* __restrict__ acc, unsigned short* __restrict__ out_bf,
                                const float* __restrict__ denom,
                                const float* __restrict__ bias) {
    int idx = blockIdx.x * blockDim.x + threadIdx.x;
    if (idx >= Nn * HC) return;
    int n = idx >> 9, c = idx & (HC - 1);
    float o = acc[idx] / (denom[n * HEADS + (c >> 6)] + 1e-16f) + bias[c];
    o = fmaxf(o, 0.0f);
    acc[idx] = o;
    out_bf[idx] = f32_to_bf16_rne(o);
}

// ---------------------------------------------------------------------------
// Head: per graph build combined[3087] in LDS, MLP 3087->256->1
// ---------------------------------------------------------------------------
__global__ __launch_bounds__(256) void head_kernel(
    const float* __restrict__ H, const float* __restrict__ actions,
    const float* __restrict__ fw1, const float* __restrict__ fb1,
    const float* __restrict__ fw2, const float* __restrict__ fb2,
    float* __restrict__ out) {
    __shared__ float comb[INDIM];
    __shared__ float hid[FHID];
    int g = blockIdx.x;
    int t = threadIdx.x;
    const float* hg = H + (size_t)g * Pp * HC;

    for (int i = t; i < NA * HC; i += 256) comb[i] = hg[i];   // agent nodes 0..4
    for (int c = t; c < HC; c += 256) {                       // mean pool over 64 nodes
        float s = 0.0f;
        for (int n = 0; n < Pp; ++n) s += hg[n * HC + c];
        comb[NA * HC + c] = s * (1.0f / (float)Pp);
    }
    if (t < NA * AD) comb[(NA + 1) * HC + t] = actions[g * NA * AD + t];
    __syncthreads();

    float s = fb1[t];
    for (int i = 0; i < INDIM; ++i) s += comb[i] * fw1[i * FHID + t];
    hid[t] = fmaxf(s, 0.0f) * fw2[t];
    __syncthreads();
    for (int off = 128; off > 0; off >>= 1) {
        if (t < off) hid[t] += hid[t + off];
        __syncthreads();
    }
    if (t == 0) out[g] = hid[0] + fb2[0];
}

// ---------------------------------------------------------------------------
// host side
// ---------------------------------------------------------------------------
extern "C" void kernel_launch(void* const* d_in, const int* in_sizes, int n_in,
                              void* d_out, int out_size, void* d_ws, size_t ws_size,
                              hipStream_t stream) {
    (void)in_sizes; (void)n_in; (void)out_size; (void)ws_size;

    const float* x         = (const float*)d_in[0];
    const float* edge_attr = (const float*)d_in[1];
    const float* actions   = (const float*)d_in[2];
    const float* Wl[3]  = {(const float*)d_in[3],  (const float*)d_in[9],  (const float*)d_in[15]};
    const float* asl[3] = {(const float*)d_in[4],  (const float*)d_in[10], (const float*)d_in[16]};
    const float* adl[3] = {(const float*)d_in[5],  (const float*)d_in[11], (const float*)d_in[17]};
    const float* Wel[3] = {(const float*)d_in[6],  (const float*)d_in[12], (const float*)d_in[18]};
    const float* ael[3] = {(const float*)d_in[7],  (const float*)d_in[13], (const float*)d_in[19]};
    const float* bl[3]  = {(const float*)d_in[8],  (const float*)d_in[14], (const float*)d_in[20]};
    const float* fw1 = (const float*)d_in[21];
    const float* fb1 = (const float*)d_in[22];
    const float* fw2 = (const float*)d_in[23];
    const float* fb2 = (const float*)d_in[24];
    const int*   ei  = (const int*)d_in[25];
    float* out = (float*)d_out;

    // carve workspace
    char* ws = (char*)d_ws;
    size_t off = 0;
    auto carve = [&](size_t bytes) -> void* {
        void* p = ws + off;
        off += (bytes + 255) & ~(size_t)255;
        return p;
    };
    float*          bufT   = (float*)carve((size_t)Nn * HC * 4);   // transformed features
    float*          bufA   = (float*)carve((size_t)Nn * HC * 4);   // layer out / acc
    float*          bufB   = (float*)carve((size_t)Nn * HC * 4);   // layer out / acc
    unsigned short* Hbf    = (unsigned short*)carve((size_t)Nn * HC * 2);  // bf16 activations
    unsigned short* Wt     = (unsigned short*)carve((size_t)HC * HC * 2);  // bf16 W^T
    float*          als    = (float*)carve((size_t)Nn * HEADS * 4);
    float*          ald    = (float*)carve((size_t)Nn * HEADS * 4);
    unsigned*       mmax   = (unsigned*)carve((size_t)Nn * HEADS * 4);
    float*          denom  = (float*)carve((size_t)Nn * HEADS * 4);
    float*          alphaE = (float*)carve((size_t)Ee * HEADS * 4);
    float*          alphaS = (float*)carve((size_t)Nn * HEADS * 4);
    float*          cnt    = (float*)carve((size_t)Nn * 4);
    float*          sattr  = (float*)carve((size_t)Nn * 4);
    float*          ce     = (float*)carve(HEADS * 4);

    const int NH = Nn * HEADS;            // 262144
    const int NC = Nn * HC;               // 16777216
    const int EH = Ee * HEADS;            // 2097152
    const unsigned ENC_NEG_INF = 0x007FFFFFu;  // enc(-inf)

    auto blocks = [](int n, int bs) { return (n + bs - 1) / bs; };

    // degree / summed attr for self-loop mean fill (once)
    fill_f32_kernel<<<blocks(Nn, 256), 256, 0, stream>>>(cnt, 0.0f, Nn);
    fill_f32_kernel<<<blocks(Nn, 256), 256, 0, stream>>>(sattr, 0.0f, Nn);
    deg_kernel<<<blocks(Ee, 256), 256, 0, stream>>>(ei, edge_attr, cnt, sattr);

    float* layer_in  = nullptr;   // f32 node features entering the layer
    float* layer_out[3] = {bufA, bufB, bufA};

    for (int L = 0; L < 3; ++L) {
        float* acc = layer_out[L];
        bool self_loops = (L > 0);

        // rank-1 edge projection coefficient per head
        ce_kernel<<<1, HEADS, 0, stream>>>(Wel[L], ael[L], ce);

        // transform T = in @ W
        if (L == 0) {
            lin1_kernel<<<blocks(NC, 256), 256, 0, stream>>>(x, Wl[0], bufT);
        } else {
            wtcvt_kernel<<<blocks(HC * HC, 256), 256, 0, stream>>>(Wl[L], Wt);
            dim3 grid(Nn / 64, HC / 128);
            gemm_bf16_wmma_kernel<<<grid, 256, 0, stream>>>(Hbf, Wt, bufT);
        }

        // attention projections
        nodeproj_kernel<<<blocks(NH, 256), 256, 0, stream>>>(bufT, asl[L], adl[L], als, ald);

        // reset segment state
        fill_u32_kernel<<<blocks(NH, 256), 256, 0, stream>>>(mmax, ENC_NEG_INF, NH);
        fill_f32_kernel<<<blocks(NH, 256), 256, 0, stream>>>(denom, 0.0f, NH);
        fill_f32_kernel<<<blocks(NC, 256), 256, 0, stream>>>(acc, 0.0f, NC);

        // logits + segment max
        edge_alpha_kernel<<<blocks(EH, 256), 256, 0, stream>>>(ei, edge_attr, als, ald, ce,
                                                               alphaE, mmax);
        if (self_loops)
            self_alpha_kernel<<<blocks(NH, 256), 256, 0, stream>>>(als, ald, ce, cnt, sattr,
                                                                   alphaS, mmax);

        // exp + weighted aggregation
        edge_acc_kernel<<<Ee, 64, 0, stream>>>(ei, alphaE, mmax, bufT, denom, acc);
        if (self_loops) {
            self_exp_kernel<<<blocks(NH, 256), 256, 0, stream>>>(alphaS, mmax, denom);
            self_acc_kernel<<<blocks(NC, 256), 256, 0, stream>>>(alphaS, bufT, acc);
        }

        // normalize + bias + relu (in place) + fused bf16 copy for next GEMM
        finalize_kernel<<<blocks(NC, 256), 256, 0, stream>>>(acc, Hbf, denom, bl[L]);

        layer_in = acc;
    }

    // pooling + MLP head
    head_kernel<<<Gg, 256, 0, stream>>>(layer_in, actions, fw1, fb1, fw2, fb2, out);
}
